// AttentionModule_56264071577844
// MI455X (gfx1250) — matrix operand order repaired and checked by reading, not compiled, not run on hardware
//
#include <hip/hip_runtime.h>

typedef __attribute__((ext_vector_type(16))) __bf16 v16bf;
typedef __attribute__((ext_vector_type(8)))  float  v8f;
typedef __attribute__((ext_vector_type(4)))  unsigned int v4u;
typedef __attribute__((ext_vector_type(8)))  int v8i;
typedef __attribute__((ext_vector_type(4)))  int v4i;

#define DIM   1024
#define SEQ   2048
#define BATCH 4
#define LDSROW 40   // 32 bf16 payload + 8 bf16 pad => 80B row stride (bank-conflict-free)

#if defined(__has_builtin)
#if __has_builtin(__builtin_amdgcn_tensor_load_to_lds) && __has_builtin(__builtin_amdgcn_s_wait_tensorcnt)
#define USE_TDM 1
#endif
#endif

__device__ __forceinline__ unsigned short f2bf(float f) {
    unsigned int u = __builtin_bit_cast(unsigned int, f);
    u = (u + 0x7FFFu + ((u >> 16) & 1u)) >> 16;   // round-to-nearest-even
    return (unsigned short)u;
}

union FragBits { uint4 q[2]; v16bf v; };

// ---- LDS fragment readers (tile rows padded to LDSROW ushorts = 80B) ----
// A-frag (16x32): lanes 0-15: row=l, K {0..7,16..23}; lanes 16-31: row=l, K {8..15,24..31}
__device__ __forceinline__ v16bf lds_frag_a(const unsigned short* row0) {
    const unsigned lane = threadIdx.x & 31u;
    const unsigned l = lane & 15u, h = lane >> 4;
    const unsigned short* p = row0 + (size_t)l * LDSROW + h * 8u;
    FragBits u;
    u.q[0] = *(const uint4*)(p);
    u.q[1] = *(const uint4*)(p + 16);
    return u.v;
}
// B-frag (32x16) from B^T rows: lanes 0-15: col=l, K {0..7,16..23}; lanes 16-31: col=l, K {8..15,24..31}
__device__ __forceinline__ v16bf lds_frag_b(const unsigned short* row0) {
    const unsigned lane = threadIdx.x & 31u;
    const unsigned l = lane & 15u, h = lane >> 4;
    const unsigned short* p = row0 + (size_t)l * LDSROW + h * 8u;
    FragBits u;
    u.q[0] = *(const uint4*)(p);
    u.q[1] = *(const uint4*)(p + 16);
    return u.v;
}

#ifdef USE_TDM
typedef __attribute__((address_space(3))) unsigned short lds_us_t;
__device__ __forceinline__ unsigned lds_byte_offset(unsigned short* p) {
    lds_us_t* p3 = (lds_us_t*)p;            // addrspacecast generic -> LDS
    return __builtin_bit_cast(unsigned, p3); // AS3 pointers are 32-bit
}

// TDM: DMA one 128(row) x 32(k) bf16 tile from global (row stride = ld elements)
// into LDS with +16B padding after every 64B row => 80B LDS row stride.
__device__ __forceinline__ void tdm_load_tile(unsigned short* ldsDst,
                                              const unsigned short* gsrc, int ld) {
    const unsigned long long ga = (unsigned long long)(uintptr_t)gsrc;
    v4u g0;
    g0[0] = 1u;                                             // count=1, user descriptor
    g0[1] = lds_byte_offset(ldsDst);                        // lds_addr (bytes)
    g0[2] = (unsigned)(ga & 0xFFFFFFFFull);                 // global_addr[31:0]
    g0[3] = (unsigned)((ga >> 32) & 0x01FFFFFFull)          // global_addr[56:32]
          | (2u << 30);                                     // type=2 ("image")
    v8i g1;
    g1[0] = (int)((1u << 16)    // data_size = 1 (2 bytes)
          |       (1u << 20)    // pad_enable
          |       (3u << 22)    // pad_interval: every 64B
          |       (3u << 25));  // pad_amount: 4 DWORDs = 16B
    g1[1] = (int)(32u << 16);   // tensor_dim0[15:0] = 32
    g1[2] = (int)(128u << 16);  // tensor_dim0 hi | tensor_dim1[15:0] = 128
    g1[3] = (int)(32u << 16);   // tensor_dim1 hi | tile_dim0 = 32
    g1[4] = (int)128;           // tile_dim1 = 128 | tile_dim2 = 0
    g1[5] = ld;                 // tensor_dim0_stride[31:0] (elements)
    g1[6] = 0;
    g1[7] = 0;
    v4i z4{};
    v8i z8{};
    __builtin_amdgcn_tensor_load_to_lds(g0, g1, z4, z4, z8, 0);  // 6-arg toolchain form
}
#else
// Fallback staging: 256 threads copy 128 rows x 32 bf16 (2 threads / row).
__device__ __forceinline__ void stage_tile_manual(unsigned short* dst,
                                                  const unsigned short* gsrc, int ld) {
    const int t = threadIdx.x;
    const int row = t >> 1, half = t & 1;
    const unsigned short* s = gsrc + (size_t)row * ld + half * 16;
    uint4 q0 = *(const uint4*)(s);
    uint4 q1 = *(const uint4*)(s + 8);
    unsigned short* d = dst + row * LDSROW + half * 16;
    *(uint4*)(d) = q0;
    *(uint4*)(d + 8) = q1;
}
#endif

// D = A * B^T (+bias), optional fp32/bf16 output, optional transposed store.
// Workgroup: 256 threads = 8 waves; WG tile 128(M) x 128(N); wave tile 32x64.
// Operand tiles staged in LDS (TDM double-buffered when available).
template<bool OUT_BF16, bool TRANS_OUT, bool HAS_BIAS>
__global__ __launch_bounds__(256) void gemm_bf16_wmma(
    const unsigned short* __restrict__ A,
    const unsigned short* __restrict__ Bt,
    const float* __restrict__ bias,
    void* __restrict__ outp,
    int K, int ldA, int ldB, int ldO,
    long long sA, long long sB, long long sO,
    float scale)
{
    __shared__ __align__(16) unsigned short smA[2][128 * LDSROW];
    __shared__ __align__(16) unsigned short smB[2][128 * LDSROW];

    const int bz = blockIdx.z;
    A  += (long long)bz * sA;
    Bt += (long long)bz * sB;

    const int m0blk = blockIdx.y * 128;
    const int n0blk = blockIdx.x * 128;
    const int mrel = (int)((threadIdx.x >> 5) & 3u) * 32;  // wave M offset (2 frags)
    const int nrel = (int)(threadIdx.x >> 7) * 64;         // wave N offset (4 frags)

    const unsigned short* Abase = A + (size_t)m0blk * (size_t)ldA;
    const unsigned short* Bbase = Bt + (size_t)n0blk * (size_t)ldB;

    v8f acc[2][4];
    #pragma unroll
    for (int i = 0; i < 2; ++i)
        #pragma unroll
        for (int j = 0; j < 4; ++j) { v8f z{}; acc[i][j] = z; }

    const int nsteps = K >> 5;

#ifdef USE_TDM
    if (threadIdx.x < 32) {                 // wave 0 drives the DMA
        tdm_load_tile(&smA[0][0], Abase, ldA);
        tdm_load_tile(&smB[0][0], Bbase, ldB);
    }
#endif

    for (int s = 0; s < nsteps; ++s) {
        const int k0 = s << 5;
#ifdef USE_TDM
        if (threadIdx.x < 32) {
            if (s + 1 < nsteps) {
                tdm_load_tile(&smA[(s + 1) & 1][0], Abase + (k0 + 32), ldA);
                tdm_load_tile(&smB[(s + 1) & 1][0], Bbase + (k0 + 32), ldB);
                __builtin_amdgcn_s_wait_tensorcnt(2);  // previous pair complete
            } else {
                __builtin_amdgcn_s_wait_tensorcnt(0);
            }
        }
        __syncthreads();
        const unsigned short* bufA = &smA[s & 1][0];
        const unsigned short* bufB = &smB[s & 1][0];
#else
        stage_tile_manual(&smA[0][0], Abase + k0, ldA);
        stage_tile_manual(&smB[0][0], Bbase + k0, ldB);
        __syncthreads();
        const unsigned short* bufA = &smA[0][0];
        const unsigned short* bufB = &smB[0][0];
#endif
        v16bf a0 = lds_frag_a(bufA + (size_t)(mrel)      * LDSROW);
        v16bf a1 = lds_frag_a(bufA + (size_t)(mrel + 16) * LDSROW);
        v16bf b0 = lds_frag_b(bufB + (size_t)(nrel +  0) * LDSROW);
        v16bf b1 = lds_frag_b(bufB + (size_t)(nrel + 16) * LDSROW);
        v16bf b2 = lds_frag_b(bufB + (size_t)(nrel + 32) * LDSROW);
        v16bf b3 = lds_frag_b(bufB + (size_t)(nrel + 48) * LDSROW);

        acc[0][0] = __builtin_amdgcn_wmma_f32_16x16x32_bf16(false, a0, false, b0, (short)0, acc[0][0], false, false);
        acc[0][1] = __builtin_amdgcn_wmma_f32_16x16x32_bf16(false, a0, false, b1, (short)0, acc[0][1], false, false);
        acc[0][2] = __builtin_amdgcn_wmma_f32_16x16x32_bf16(false, a0, false, b2, (short)0, acc[0][2], false, false);
        acc[0][3] = __builtin_amdgcn_wmma_f32_16x16x32_bf16(false, a0, false, b3, (short)0, acc[0][3], false, false);
        acc[1][0] = __builtin_amdgcn_wmma_f32_16x16x32_bf16(false, a1, false, b0, (short)0, acc[1][0], false, false);
        acc[1][1] = __builtin_amdgcn_wmma_f32_16x16x32_bf16(false, a1, false, b1, (short)0, acc[1][1], false, false);
        acc[1][2] = __builtin_amdgcn_wmma_f32_16x16x32_bf16(false, a1, false, b2, (short)0, acc[1][2], false, false);
        acc[1][3] = __builtin_amdgcn_wmma_f32_16x16x32_bf16(false, a1, false, b3, (short)0, acc[1][3], false, false);
        __syncthreads();   // tile consumed before DMA overwrites it
    }

    // C/D layout: lanes 0-15 => N=lane, M=vgpr r; lanes 16-31 => N=lane-16, M=8+r
    const unsigned lane = threadIdx.x & 31u;
    const int l = (int)(lane & 15u), h = (int)(lane >> 4);
    const int m0 = m0blk + mrel;
    const int n0 = n0blk + nrel;
    #pragma unroll
    for (int i = 0; i < 2; ++i) {
        #pragma unroll
        for (int j = 0; j < 4; ++j) {
            const int nn = n0 + 16 * j + l;
            const float bv = HAS_BIAS ? bias[nn] : 0.0f;
            #pragma unroll
            for (int r = 0; r < 8; ++r) {
                const int mm = m0 + 16 * i + r + 8 * h;
                const float val = acc[i][j][r] * scale + bv;
                if (OUT_BF16) {
                    unsigned short* o = (unsigned short*)outp + (long long)bz * sO;
                    if (TRANS_OUT) o[(size_t)nn * (size_t)ldO + mm] = f2bf(val);
                    else           o[(size_t)mm * (size_t)ldO + nn] = f2bf(val);
                } else {
                    float* o = (float*)outp + (long long)bz * sO;
                    o[(size_t)mm * (size_t)ldO + nn] = val;
                }
            }
        }
    }
}

// Row softmax over SEQ fp32 logits -> bf16 probabilities. One block per row.
__global__ __launch_bounds__(256) void softmax_rows_bf16(
    const float* __restrict__ S, unsigned short* __restrict__ P)
{
    const int row = blockIdx.x;
    const float4* s = (const float4*)(S + (size_t)row * SEQ);
    const int t = threadIdx.x;

    float4 x0 = s[t];
    float4 x1 = s[t + 256];

    float mx = fmaxf(fmaxf(fmaxf(x0.x, x0.y), fmaxf(x0.z, x0.w)),
                     fmaxf(fmaxf(x1.x, x1.y), fmaxf(x1.z, x1.w)));
    #pragma unroll
    for (int o = 16; o > 0; o >>= 1) mx = fmaxf(mx, __shfl_xor(mx, o, 32));

    __shared__ float sred[8];
    const int wid = t >> 5, lane = t & 31;
    if (lane == 0) sred[wid] = mx;
    __syncthreads();
    float rmax = sred[0];
    #pragma unroll
    for (int i = 1; i < 8; ++i) rmax = fmaxf(rmax, sred[i]);
    __syncthreads();

    float e[8];
    e[0] = __expf(x0.x - rmax); e[1] = __expf(x0.y - rmax);
    e[2] = __expf(x0.z - rmax); e[3] = __expf(x0.w - rmax);
    e[4] = __expf(x1.x - rmax); e[5] = __expf(x1.y - rmax);
    e[6] = __expf(x1.z - rmax); e[7] = __expf(x1.w - rmax);
    float sum = ((e[0] + e[1]) + (e[2] + e[3])) + ((e[4] + e[5]) + (e[6] + e[7]));
    #pragma unroll
    for (int o = 16; o > 0; o >>= 1) sum += __shfl_xor(sum, o, 32);
    if (lane == 0) sred[wid] = sum;
    __syncthreads();
    float tot = 0.0f;
    #pragma unroll
    for (int i = 0; i < 8; ++i) tot += sred[i];
    const float inv = 1.0f / tot;

    uint2 o0, o1;
    o0.x = (unsigned)f2bf(e[0] * inv) | ((unsigned)f2bf(e[1] * inv) << 16);
    o0.y = (unsigned)f2bf(e[2] * inv) | ((unsigned)f2bf(e[3] * inv) << 16);
    o1.x = (unsigned)f2bf(e[4] * inv) | ((unsigned)f2bf(e[5] * inv) << 16);
    o1.y = (unsigned)f2bf(e[6] * inv) | ((unsigned)f2bf(e[7] * inv) << 16);
    uint2* p = (uint2*)(P + (size_t)row * SEQ);
    p[t] = o0;
    p[t + 256] = o1;
}

__global__ __launch_bounds__(256) void cast_f32_to_bf16(
    const float* __restrict__ in, unsigned short* __restrict__ out, int n4)
{
    const int i = blockIdx.x * 256 + threadIdx.x;
    if (i >= n4) return;
    float4 f = ((const float4*)in)[i];
    uint2 o;
    o.x = (unsigned)f2bf(f.x) | ((unsigned)f2bf(f.y) << 16);
    o.y = (unsigned)f2bf(f.z) | ((unsigned)f2bf(f.w) << 16);
    ((uint2*)out)[i] = o;
}

extern "C" void kernel_launch(void* const* d_in, const int* in_sizes, int n_in,
                              void* d_out, int out_size, void* d_ws, size_t ws_size,
                              hipStream_t stream)
{
    (void)in_sizes; (void)n_in; (void)out_size; (void)ws_size;

    const float* x  = (const float*)d_in[0];
    const float* Wq = (const float*)d_in[1];
    const float* bq = (const float*)d_in[2];
    const float* Wk = (const float*)d_in[3];
    const float* bk = (const float*)d_in[4];
    const float* Wv = (const float*)d_in[5];
    const float* bv = (const float*)d_in[6];

    char* ws = (char*)d_ws;
    size_t off = 0;
    auto carve = [&](size_t bytes) -> void* {
        void* p = (void*)(ws + off);
        off += (bytes + 255) & ~(size_t)255;
        return p;
    };
    unsigned short* xb  = (unsigned short*)carve((size_t)BATCH * SEQ * DIM * 2);
    unsigned short* Wqb = (unsigned short*)carve((size_t)DIM * DIM * 2);
    unsigned short* Wkb = (unsigned short*)carve((size_t)DIM * DIM * 2);
    unsigned short* Wvb = (unsigned short*)carve((size_t)DIM * DIM * 2);
    unsigned short* qb  = (unsigned short*)carve((size_t)BATCH * SEQ * DIM * 2);
    unsigned short* kb  = (unsigned short*)carve((size_t)BATCH * SEQ * DIM * 2);
    unsigned short* vT  = (unsigned short*)carve((size_t)BATCH * DIM * SEQ * 2);  // [b][d][j]
    float*          S   = (float*)carve((size_t)BATCH * SEQ * SEQ * 4);
    unsigned short* P   = (unsigned short*)carve((size_t)BATCH * SEQ * SEQ * 2);

    // 1) casts
    cast_f32_to_bf16<<<(BATCH * SEQ * DIM / 4 + 255) / 256, 256, 0, stream>>>(x, xb, BATCH * SEQ * DIM / 4);
    cast_f32_to_bf16<<<(DIM * DIM / 4 + 255) / 256, 256, 0, stream>>>(Wq, Wqb, DIM * DIM / 4);
    cast_f32_to_bf16<<<(DIM * DIM / 4 + 255) / 256, 256, 0, stream>>>(Wk, Wkb, DIM * DIM / 4);
    cast_f32_to_bf16<<<(DIM * DIM / 4 + 255) / 256, 256, 0, stream>>>(Wv, Wvb, DIM * DIM / 4);

    const dim3 blk(256);
    const dim3 gProj(DIM / 128, SEQ / 128, BATCH);   // (8,16,4)
    const dim3 gScore(SEQ / 128, SEQ / 128, BATCH);  // (16,16,4)
    const dim3 gOut(DIM / 128, SEQ / 128, BATCH);    // (8,16,4)
    const long long sTok = (long long)SEQ * DIM;     // per-batch token-block stride
    const long long sSS  = (long long)SEQ * SEQ;

    // 2) q = x Wq^T + bq ; k = x Wk^T + bk (bf16 out)
    gemm_bf16_wmma<true, false, true><<<gProj, blk, 0, stream>>>(
        xb, Wqb, bq, qb, DIM, DIM, DIM, DIM, sTok, 0, sTok, 1.0f);
    gemm_bf16_wmma<true, false, true><<<gProj, blk, 0, stream>>>(
        xb, Wkb, bk, kb, DIM, DIM, DIM, DIM, sTok, 0, sTok, 1.0f);
    // 3) v = x Wv^T + bv, stored transposed per batch: vT[d][j]
    gemm_bf16_wmma<true, true, true><<<gProj, blk, 0, stream>>>(
        xb, Wvb, bv, vT, DIM, DIM, DIM, SEQ, sTok, 0, sTok, 1.0f);
    // 4) S = (q k^T) * C^-0.5  (fp32 out, batched)
    gemm_bf16_wmma<false, false, false><<<gScore, blk, 0, stream>>>(
        qb, kb, nullptr, S, DIM, DIM, DIM, SEQ, sTok, sTok, sSS, 0.03125f);
    // 5) P = softmax(S) (bf16 out)
    softmax_rows_bf16<<<BATCH * SEQ, 256, 0, stream>>>(S, P);
    // 6) out = P v  (via vT, fp32 out to d_out)
    gemm_bf16_wmma<false, false, false><<<gOut, blk, 0, stream>>>(
        P, vT, nullptr, (float*)d_out, SEQ, SEQ, SEQ, DIM, sSS, (long long)DIM * SEQ, sTok, 1.0f);
}
